// EnhancedContextAwareDualVQ_24902220382616
// MI455X (gfx1250) — matrix operand order repaired and checked by reading, not compiled, not run on hardware
//
#include <hip/hip_runtime.h>

// ---------------------------------------------------------------------------
// Problem constants (match reference)
// ---------------------------------------------------------------------------
#define N_TOK   131072
#define LATENT  128
#define DIM     256
#define NSYN    64
#define NSEM    128

#define TPB     64      // tokens per block
#define THREADS 128     // 4 waves of 32
#define LDA     264     // bf16 row stride for z/h1 tile (8-elem pad -> conflict-free b128)
#define LDF     260     // fp32 row stride for GEMM1 staging
#define LDH2    136     // bf16 row stride for h2 tile
#define LDSC    132     // fp32 row stride for scores

typedef __attribute__((ext_vector_type(16))) __bf16 v16bf;
typedef __attribute__((ext_vector_type(8)))  __bf16 v8bf;
typedef __attribute__((ext_vector_type(4)))  __bf16 v4bf;
typedef __attribute__((ext_vector_type(8)))  float  v8f;
typedef __attribute__((ext_vector_type(4)))  unsigned int v4u;
typedef __attribute__((ext_vector_type(8)))  int    v8i;
typedef __attribute__((ext_vector_type(4)))  int    v4i;

#if __has_builtin(__builtin_amdgcn_tensor_load_to_lds) && __has_builtin(__builtin_amdgcn_s_wait_tensorcnt)
#define USE_TDM 1
#else
#define USE_TDM 0
#endif

static __device__ __forceinline__ v16bf cat8(v8bf lo, v8bf hi) {
    return __builtin_shufflevector(lo, hi, 0,1,2,3,4,5,6,7,8,9,10,11,12,13,14,15);
}

// A-fragment (16x32 bf16) from LDS, row-major tile with row stride lda.
static __device__ __forceinline__ v16bf load_afrag(const __bf16* src, int lda,
                                                   int mbase, int kb, int lane) {
    int m  = mbase + (lane & 15);
    int hl = lane >> 4;
    const __bf16* p = src + m * lda + kb + hl * 8;
    v8bf lo = *(const v8bf*)p;          // K = kb + hl*8 + [0..7]
    v8bf hi = *(const v8bf*)(p + 16);   // K = kb + 16 + hl*8 + [0..7]
    return cat8(lo, hi);
}

// B-fragment (32x16 bf16) from global weights pre-transposed to [n][k] bf16.
static __device__ __forceinline__ v16bf load_bfrag(const __bf16* wt, int ldk,
                                                   int nbase, int kb, int lane) {
    int n  = nbase + (lane & 15);
    int hl = lane >> 4;
    const __bf16* p = wt + n * ldk + kb + hl * 8;
    v8bf lo = *(const v8bf*)p;
    v8bf hi = *(const v8bf*)(p + 16);
    return cat8(lo, hi);
}

#define WMMA_BF16(a, b, c) \
    __builtin_amdgcn_wmma_f32_16x16x32_bf16(false, (a), false, (b), (short)0, (c), false, false)

// ---------------------------------------------------------------------------
// Prep kernels
// ---------------------------------------------------------------------------
__global__ void k_zero(float* p, int n) {
    int i = blockIdx.x * blockDim.x + threadIdx.x;
    if (i < n) p[i] = 0.0f;
}

// dst[o*rows_in + r] = bf16(src[r*cols_out + o])   (weights -> [out][in] bf16)
__global__ void k_transpose_bf16(const float* __restrict__ src, __bf16* __restrict__ dst,
                                 int rows_in, int cols_out) {
    int i = blockIdx.x * blockDim.x + threadIdx.x;
    if (i < rows_in * cols_out) {
        int o = i % cols_out;
        int r = i / cols_out;
        dst[o * rows_in + r] = (__bf16)src[i];
    }
}

// codebook: bf16 copy (already [code][dim] = B-friendly) + fp32 row norms
__global__ __launch_bounds__(256) void k_cbprep(const float* __restrict__ cb,
                                                __bf16* __restrict__ cbbf,
                                                float* __restrict__ cbnorm) {
    __shared__ float red[256];
    int k = blockIdx.x, d = threadIdx.x;
    float v = cb[k * DIM + d];
    cbbf[k * DIM + d] = (__bf16)v;
    red[d] = v * v;
    __syncthreads();
    for (int s = 128; s > 0; s >>= 1) { if (d < s) red[d] += red[d + s]; __syncthreads(); }
    if (d == 0) cbnorm[k] = red[0];
}

// ---------------------------------------------------------------------------
// Main fused VQ kernel (one block = 64 tokens, blockIdx.y selects stream)
// ---------------------------------------------------------------------------
struct VQArgs {
    const float* z_re;  const float* z_im;
    const int*   prev;
    const float* cb;      // fp32 [K][256]
    const float* adj;     // fp32 [K][K]
    const float* b1; const float* ln_g; const float* ln_b;
    const float* b2; const float* bp;
    const __bf16* W1t;    // [256][256]  ([out][in])
    const __bf16* W2t;    // [128][256]
    const __bf16* Wpt;    // [K][128]
    const __bf16* cbbf;   // [K][256]
    const float* cbnorm;  // [K]
    float* counts;        // [K]
    float* esum;          // [K][256]
    float* out_q;         // [N][256]
    float* out_idx;       // [N]
    float* out_loss;      // [1]
    int K;
};

__global__ __launch_bounds__(THREADS) void vq_main(VQArgs syn, VQArgs sem) {
    VQArgs A = (blockIdx.y == 0) ? syn : sem;

    __shared__ __align__(16) __bf16 sZ [TPB * LDA];   // z bf16, later h1 bf16
    __shared__ __align__(16) float  sF [TPB * LDF];   // TDM fp32 staging, then GEMM1 fp32
    __shared__ __align__(16) __bf16 sH2[TPB * LDH2];  // h2 bf16
    __shared__ __align__(16) float  sS [TPB * LDSC];  // scores
    __shared__ float  sZn[TPB];
    __shared__ int    sIdx[TPB];
    __shared__ float  sRed[THREADS];

    const int tid  = threadIdx.x;
    const int lane = tid & 31;
    const int wave = tid >> 5;
    const int mbase = wave * 16;
    const int hl   = lane >> 4;
    const int nlo  = lane & 15;
    const int t0   = blockIdx.x * TPB;
    const int nk   = A.K >> 4;

    if (tid < TPB) sZn[tid] = 0.0f;

#if USE_TDM
    // ---- Tensor Data Mover: DMA the 64x128 fp32 re/im tiles into LDS ----
    if (wave == 0) {
        unsigned lds_re = (unsigned)(uintptr_t)(void*)&sF[0];
        unsigned lds_im = lds_re + (unsigned)(TPB * LATENT * 4);
        unsigned long long ga_re = (unsigned long long)(uintptr_t)(A.z_re + (size_t)t0 * LATENT);
        unsigned long long ga_im = (unsigned long long)(uintptr_t)(A.z_im + (size_t)t0 * LATENT);
        // D# group1: data_size=4B; tensor 128x64; tile 128x64; row stride 128
        v8i g1 = { 0x00020000,            // workgroup_mask=0, data_size=2 (4B)
                   (int)(128u << 16),     // tensor_dim0=128 (bits 79:48 low part)
                   (int)(64u  << 16),     // tensor_dim1=64  (bits 111:80 low part)
                   (int)(128u << 16),     // tile_dim0=128   (bits 127:112)
                   64,                    // tile_dim1=64    (bits 143:128)
                   128,                   // tensor_dim0_stride=128 (bits 207:160)
                   0, 0 };
        v4i gz4 = { 0, 0, 0, 0 };
        v8i gz8 = { 0, 0, 0, 0, 0, 0, 0, 0 };
        v4u g0_re = { 1u, lds_re, (unsigned)ga_re,
                      (unsigned)((ga_re >> 32) & 0x01FFFFFFull) | (2u << 30) };
        v4u g0_im = { 1u, lds_im, (unsigned)ga_im,
                      (unsigned)((ga_im >> 32) & 0x01FFFFFFull) | (2u << 30) };
        __builtin_amdgcn_tensor_load_to_lds(g0_re, g1, gz4, gz4, gz8, 0);
        __builtin_amdgcn_tensor_load_to_lds(g0_im, g1, gz4, gz4, gz8, 0);
        __builtin_amdgcn_s_wait_tensorcnt(0);
    }
    __syncthreads();
    // convert staged fp32 -> padded bf16 WMMA tile + per-token squared norm
    for (int q = tid; q < TPB * (DIM / 4); q += THREADS) {
        int t = q >> 6;
        int d = (q & 63) * 4;
        const float4* zp = (d < LATENT)
            ? (const float4*)(&sF[t * LATENT + d])
            : (const float4*)(&sF[TPB * LATENT + t * LATENT + d - LATENT]);
        float4 zv = *zp;
        v4bf b; b[0] = (__bf16)zv.x; b[1] = (__bf16)zv.y;
                b[2] = (__bf16)zv.z; b[3] = (__bf16)zv.w;
        *(v4bf*)(&sZ[t * LDA + d]) = b;
        atomicAdd(&sZn[t], zv.x * zv.x + zv.y * zv.y + zv.z * zv.z + zv.w * zv.w);
    }
#else
    __syncthreads();
    for (int q = tid; q < TPB * (DIM / 4); q += THREADS) {
        int t = q >> 6;
        int d = (q & 63) * 4;
        int tok = t0 + t;
        const float4* zp = (d < LATENT)
            ? (const float4*)(A.z_re + (size_t)tok * LATENT + d)
            : (const float4*)(A.z_im + (size_t)tok * LATENT + d - LATENT);
        float4 zv = *zp;
        v4bf b; b[0] = (__bf16)zv.x; b[1] = (__bf16)zv.y;
                b[2] = (__bf16)zv.z; b[3] = (__bf16)zv.w;
        *(v4bf*)(&sZ[t * LDA + d]) = b;
        atomicAdd(&sZn[t], zv.x * zv.x + zv.y * zv.y + zv.z * zv.z + zv.w * zv.w);
    }
#endif
    __syncthreads();

    // ---- preload A fragments for the wave's 16-token strip (K = 256) ----
    v16bf af[8];
#pragma unroll
    for (int kb = 0; kb < 8; ++kb) af[kb] = load_afrag(sZ, LDA, mbase, kb * 32, lane);

    // ---- stage 1: scores = 2 * z @ cb^T ----
    for (int nt = 0; nt < nk; ++nt) {
        v8f acc = {};
#pragma unroll
        for (int kb = 0; kb < 8; ++kb) {
            v16bf b = load_bfrag(A.cbbf, DIM, nt * 16, kb * 32, lane);
            acc = WMMA_BF16(af[kb], b, acc);
        }
        int n = nt * 16 + nlo;
#pragma unroll
        for (int j = 0; j < 8; ++j)
            sS[(mbase + j + 8 * hl) * LDSC + n] = 2.0f * acc[j];
    }

    // ---- stage 2: GEMM1 h1 = z @ W1 + b1 -> fp32 staging ----
    for (int nt = 0; nt < 16; ++nt) {
        v8f acc = {};
#pragma unroll
        for (int kb = 0; kb < 8; ++kb) {
            v16bf b = load_bfrag(A.W1t, DIM, nt * 16, kb * 32, lane);
            acc = WMMA_BF16(af[kb], b, acc);
        }
        int n = nt * 16 + nlo;
        float bias = A.b1[n];
#pragma unroll
        for (int j = 0; j < 8; ++j)
            sF[(mbase + j + 8 * hl) * LDF + n] = acc[j] + bias;
    }
    __syncthreads();

    // ---- stage 3: LayerNorm + ReLU -> bf16 back into sZ ----
    if (tid < TPB) {
        float s1 = 0.0f, s2 = 0.0f;
        for (int d = 0; d < DIM; ++d) {
            float v = sF[tid * LDF + d];
            s1 += v; s2 += v * v;
        }
        float m  = s1 * (1.0f / DIM);
        float vr = s2 * (1.0f / DIM) - m * m;
        float r  = rsqrtf(vr + 1e-5f);
        for (int d = 0; d < DIM; ++d) {
            float v = (sF[tid * LDF + d] - m) * r * A.ln_g[d] + A.ln_b[d];
            sZ[tid * LDA + d] = (__bf16)fmaxf(v, 0.0f);
        }
    }
    __syncthreads();

    // ---- stage 4: GEMM2 h2 = relu(h1 @ W2 + b2) -> bf16 tile ----
#pragma unroll
    for (int kb = 0; kb < 8; ++kb) af[kb] = load_afrag(sZ, LDA, mbase, kb * 32, lane);
    for (int nt = 0; nt < 8; ++nt) {
        v8f acc = {};
#pragma unroll
        for (int kb = 0; kb < 8; ++kb) {
            v16bf b = load_bfrag(A.W2t, DIM, nt * 16, kb * 32, lane);
            acc = WMMA_BF16(af[kb], b, acc);
        }
        int n = nt * 16 + nlo;
        float bias = A.b2[n];
#pragma unroll
        for (int j = 0; j < 8; ++j)
            sH2[(mbase + j + 8 * hl) * LDH2 + n] = (__bf16)fmaxf(acc[j] + bias, 0.0f);
    }
    // same-wave LDS producer/consumer: no block barrier needed

    // ---- stage 5: GEMM3 logits = h2 @ Wp + bp; scores += 3*logits ----
    v16bf af2[4];
#pragma unroll
    for (int kb = 0; kb < 4; ++kb) af2[kb] = load_afrag(sH2, LDH2, mbase, kb * 32, lane);
    for (int nt = 0; nt < nk; ++nt) {
        v8f acc = {};
#pragma unroll
        for (int kb = 0; kb < 4; ++kb) {
            v16bf b = load_bfrag(A.Wpt, 128, nt * 16, kb * 32, lane);
            acc = WMMA_BF16(af2[kb], b, acc);
        }
        int n = nt * 16 + nlo;
        float bias = A.bp[n];
#pragma unroll
        for (int j = 0; j < 8; ++j)
            sS[(mbase + j + 8 * hl) * LDSC + n] += 3.0f * (acc[j] + bias);
    }
    __syncthreads();

    // ---- epilogue 1: argmax per token (adj bias + norm terms) ----
    if (tid < TPB) {
        int tok = t0 + tid;
        float zn = sZn[tid];
        const float* adjrow = A.adj + (long)A.prev[tok] * A.K;
        float best = -3.4e38f;
        int   bi = 0;
        for (int k = 0; k < A.K; ++k) {
            float s = sS[tid * LDSC + k] - zn - A.cbnorm[k] + 1.5f * adjrow[k];
            if (s > best) { best = s; bi = k; }
        }
        sIdx[tid] = bi;
        A.out_idx[tok] = (float)bi;
        atomicAdd(&A.counts[bi], 1.0f);
    }
    __syncthreads();

    // ---- epilogue 2: gather zq, straight-through out, loss, EMA esum ----
    float lacc = 0.0f;
    for (int q = tid; q < TPB * (DIM / 4); q += THREADS) {
        int t = q >> 6;
        int d = (q & 63) * 4;
        int tok = t0 + t;
        const float4* zp = (d < LATENT)
            ? (const float4*)(A.z_re + (size_t)tok * LATENT + d)
            : (const float4*)(A.z_im + (size_t)tok * LATENT + d - LATENT);
        float4 zv = *zp;
        int k = sIdx[t];
        float4 qv = *(const float4*)(A.cb + (size_t)k * DIM + d);
        float4 ov;
        ov.x = zv.x + (qv.x - zv.x);
        ov.y = zv.y + (qv.y - zv.y);
        ov.z = zv.z + (qv.z - zv.z);
        ov.w = zv.w + (qv.w - zv.w);
        *(float4*)(A.out_q + (size_t)tok * DIM + d) = ov;
        float dx = qv.x - zv.x, dy = qv.y - zv.y, dz = qv.z - zv.z, dw = qv.w - zv.w;
        lacc += dx * dx + dy * dy + dz * dz + dw * dw;
        atomicAdd(&A.esum[k * DIM + d + 0], zv.x);
        atomicAdd(&A.esum[k * DIM + d + 1], zv.y);
        atomicAdd(&A.esum[k * DIM + d + 2], zv.z);
        atomicAdd(&A.esum[k * DIM + d + 3], zv.w);
    }
    sRed[tid] = lacc;
    __syncthreads();
    for (int s = THREADS / 2; s > 0; s >>= 1) {
        if (tid < s) sRed[tid] += sRed[tid + s];
        __syncthreads();
    }
    if (tid == 0)
        atomicAdd(A.out_loss, sRed[0] * (1.25f / ((float)N_TOK * (float)DIM)));
}

// ---------------------------------------------------------------------------
// Finalize: EMA codebook -> normalized Gram -> ortho loss (per stream)
// ---------------------------------------------------------------------------
__global__ __launch_bounds__(256) void k_final(const float* __restrict__ cl,
                                               const float* __restrict__ avg,
                                               const float* __restrict__ counts,
                                               const float* __restrict__ esum,
                                               int K, float* loss) {
    __shared__ __bf16 cbn[NSEM * DIM];  // 64KB (syn uses first 64 rows)
    __shared__ float  sred[256];
    __shared__ float  scl[NSEM];
    __shared__ float  sbc;
    int tid = threadIdx.x;

    float part = 0.0f;
    for (int k = tid; k < K; k += 256) {
        float c = cl[k] * 0.99f + 0.01f * counts[k];
        scl[k] = c;
        part += c;
    }
    sred[tid] = part;
    __syncthreads();
    for (int s = 128; s > 0; s >>= 1) { if (tid < s) sred[tid] += sred[tid + s]; __syncthreads(); }
    if (tid == 0) sbc = sred[0];
    __syncthreads();
    float n = sbc;

    // per-row cb_new + row normalization, tid == dim index
    for (int k = 0; k < K; ++k) {
        float cs = (scl[k] + 1e-6f) / (n + (float)K * 1e-6f) * n;
        float v  = (avg[k * DIM + tid] * 0.99f + 0.01f * esum[k * DIM + tid]) / cs;
        sred[tid] = v * v;
        __syncthreads();
        for (int s = 128; s > 0; s >>= 1) { if (tid < s) sred[tid] += sred[tid + s]; __syncthreads(); }
        if (tid == 0) sbc = fmaxf(sqrtf(sred[0]), 1e-12f);
        __syncthreads();
        cbn[k * DIM + tid] = (__bf16)(v / sbc);
        __syncthreads();
    }

    // ortho = 0.001 * mean((cbn @ cbn^T - I)^2)
    float oacc = 0.0f;
    for (int p = tid; p < K * K; p += 256) {
        int i = p / K, j = p % K;
        float dot = 0.0f;
        for (int d = 0; d < DIM; ++d)
            dot += (float)cbn[i * DIM + d] * (float)cbn[j * DIM + d];
        float g = dot - ((i == j) ? 1.0f : 0.0f);
        oacc += g * g;
    }
    sred[tid] = oacc;
    __syncthreads();
    for (int s = 128; s > 0; s >>= 1) { if (tid < s) sred[tid] += sred[tid + s]; __syncthreads(); }
    if (tid == 0) atomicAdd(loss, 0.001f * sred[0] / (float)(K * K));
}

// ---------------------------------------------------------------------------
// Launch
// ---------------------------------------------------------------------------
extern "C" void kernel_launch(void* const* d_in, const int* in_sizes, int n_in,
                              void* d_out, int out_size, void* d_ws, size_t ws_size,
                              hipStream_t stream) {
    (void)in_sizes; (void)n_in; (void)out_size; (void)ws_size;

    // inputs in setup_inputs() dict order (nested dicts depth-first)
    const float* z_fr   = (const float*)d_in[0];
    const float* z_fi   = (const float*)d_in[1];
    const float* z_sr   = (const float*)d_in[2];
    const float* z_si   = (const float*)d_in[3];
    const int*   p_syn  = (const int*)  d_in[4];
    const int*   p_sem  = (const int*)  d_in[5];
    const float* cb_syn = (const float*)d_in[6];
    const float* cb_sem = (const float*)d_in[7];
    const float* g1W1 = (const float*)d_in[8];
    const float* g1b1 = (const float*)d_in[9];
    const float* g1lg = (const float*)d_in[10];
    const float* g1lb = (const float*)d_in[11];
    const float* g1W2 = (const float*)d_in[12];
    const float* g1b2 = (const float*)d_in[13];
    const float* g1Wp = (const float*)d_in[14];
    const float* g1bp = (const float*)d_in[15];
    const float* g2W1 = (const float*)d_in[16];
    const float* g2b1 = (const float*)d_in[17];
    const float* g2lg = (const float*)d_in[18];
    const float* g2lb = (const float*)d_in[19];
    const float* g2W2 = (const float*)d_in[20];
    const float* g2b2 = (const float*)d_in[21];
    const float* g2Wp = (const float*)d_in[22];
    const float* g2bp = (const float*)d_in[23];
    const float* cl_syn  = (const float*)d_in[24];
    const float* avg_syn = (const float*)d_in[25];
    const float* adj_syn = (const float*)d_in[26];
    const float* cl_sem  = (const float*)d_in[27];
    const float* avg_sem = (const float*)d_in[28];
    const float* adj_sem = (const float*)d_in[29];

    // d_out layout: out[2][N][256], loss, idx_syn[N], idx_sem[N]
    float* out  = (float*)d_out;
    const long OUT_ELEMS = 2L * N_TOK * DIM;
    float* loss_ptr = out + OUT_ELEMS;
    float* idx_syn  = loss_ptr + 1;
    float* idx_sem  = idx_syn + N_TOK;

    // workspace layout (bytes)
    char* ws = (char*)d_ws;
    __bf16* W1t_syn  = (__bf16*)(ws + 0);        // 256*256
    __bf16* W1t_sem  = (__bf16*)(ws + 131072);   // 256*256
    __bf16* W2t_syn  = (__bf16*)(ws + 262144);   // 128*256
    __bf16* W2t_sem  = (__bf16*)(ws + 327680);   // 128*256
    __bf16* Wpt_syn  = (__bf16*)(ws + 393216);   // 64*128
    __bf16* Wpt_sem  = (__bf16*)(ws + 409600);   // 128*128
    __bf16* cbbf_syn = (__bf16*)(ws + 442368);   // 64*256
    __bf16* cbbf_sem = (__bf16*)(ws + 475136);   // 128*256
    float* cbn_syn   = (float*)(ws + 540672);    // 64
    float* cbn_sem   = (float*)(ws + 540928);    // 128
    float* counts_syn = (float*)(ws + 541440);   // 64
    float* counts_sem = counts_syn + NSYN;       // 128
    float* esum_syn   = counts_sem + NSEM;       // 64*256
    float* esum_sem   = esum_syn + NSYN * DIM;   // 128*256
    const int STATS_FLOATS = NSYN + NSEM + (NSYN + NSEM) * DIM;

    // ---- prep: zero stats + loss, bf16 weight transposes, codebook prep ----
    k_zero<<<(STATS_FLOATS + 255) / 256, 256, 0, stream>>>(counts_syn, STATS_FLOATS);
    k_zero<<<1, 32, 0, stream>>>(loss_ptr, 1);
    k_transpose_bf16<<<256, 256, 0, stream>>>(g1W1, W1t_syn, 256, 256);
    k_transpose_bf16<<<256, 256, 0, stream>>>(g2W1, W1t_sem, 256, 256);
    k_transpose_bf16<<<128, 256, 0, stream>>>(g1W2, W2t_syn, 256, 128);
    k_transpose_bf16<<<128, 256, 0, stream>>>(g2W2, W2t_sem, 256, 128);
    k_transpose_bf16<<< 32, 256, 0, stream>>>(g1Wp, Wpt_syn, 128, NSYN);
    k_transpose_bf16<<< 64, 256, 0, stream>>>(g2Wp, Wpt_sem, 128, NSEM);
    k_cbprep<<<NSYN, 256, 0, stream>>>(cb_syn, cbbf_syn, cbn_syn);
    k_cbprep<<<NSEM, 256, 0, stream>>>(cb_sem, cbbf_sem, cbn_sem);

    // ---- main fused VQ (both streams) ----
    VQArgs a_syn = { z_fr, z_fi, p_syn, cb_syn, adj_syn,
                     g1b1, g1lg, g1lb, g1b2, g1bp,
                     W1t_syn, W2t_syn, Wpt_syn, cbbf_syn, cbn_syn,
                     counts_syn, esum_syn,
                     out, idx_syn, loss_ptr, NSYN };
    VQArgs a_sem = { z_sr, z_si, p_sem, cb_sem, adj_sem,
                     g2b1, g2lg, g2lb, g2b2, g2bp,
                     W1t_sem, W2t_sem, Wpt_sem, cbbf_sem, cbn_sem,
                     counts_sem, esum_sem,
                     out + (long)N_TOK * DIM, idx_sem, loss_ptr, NSEM };
    dim3 grid(N_TOK / TPB, 2);
    vq_main<<<grid, THREADS, 0, stream>>>(a_syn, a_sem);

    // ---- finalize: EMA codebook + ortho loss ----
    k_final<<<1, 256, 0, stream>>>(cl_syn, avg_syn, counts_syn, esum_syn, NSYN, loss_ptr);
    k_final<<<1, 256, 0, stream>>>(cl_sem, avg_sem, counts_sem, esum_sem, NSEM, loss_ptr);
}
